// MultiHeadAttention_31447750541581
// MI455X (gfx1250) — matrix operand order repaired and checked by reading, not compiled, not run on hardware
//
#include <hip/hip_runtime.h>
#include <hip/hip_bf16.h>
#include <stdint.h>

#define BB 4
#define SS 1024
#define DDIM 1024
#define HH 16
#define HD 64

typedef __attribute__((ext_vector_type(16))) __bf16 v16bf;
typedef __attribute__((ext_vector_type(8)))  __bf16 v8bf;
typedef __attribute__((ext_vector_type(8)))  float  v8f;
typedef unsigned int u32;
typedef unsigned long long u64;
typedef __attribute__((ext_vector_type(4))) u32 u32x4;
typedef __attribute__((ext_vector_type(8))) u32 u32x8;

union AFrag  { v16bf v; v8bf h[2]; };
union BFragU { v16bf v; __bf16 e[16]; };

static __device__ __forceinline__ v8f wmma_bf16(v16bf a, v16bf b, v8f c) {
  // D(16x16,f32) = A(16x32,bf16) * B(32x16,bf16) + C
  return __builtin_amdgcn_wmma_f32_16x16x32_bf16(false, a, false, b, (short)0, c,
                                                 false, false);
}

// TDM: 2D tile (tile_d1 rows x tile_d0 elems of 2 bytes) Global -> LDS.
// D# bit layout per CDNA5 ISA 8.3/8.4 (group0: count/lds/global/type,
// group1: data_size, tensor dims, tile dims, dim0 stride).
static __device__ __forceinline__ void tdm_load_2d_bf16(
    u32 lds_byte_addr, u64 global_byte_addr,
    u32 tensor_d0, u32 tensor_d1, u32 tile_d0, u32 tile_d1, u64 stride0_elems) {
  u32x4 g0;
  g0.x = 1u;                                                    // count=1
  g0.y = lds_byte_addr;                                         // lds_addr
  g0.z = (u32)global_byte_addr;                                 // global[31:0]
  g0.w = ((u32)(global_byte_addr >> 32) & 0x01FFFFFFu)          // global[56:32]
         | (2u << 30);                                          // type=2 (image)
  u32x8 g1;
  g1.s0 = 1u << 16;                                             // data_size=1: 2B
  g1.s1 = (tensor_d0 & 0xFFFFu) << 16;                          // tensor_dim0 lo
  g1.s2 = (tensor_d0 >> 16) | ((tensor_d1 & 0xFFFFu) << 16);    // dim0 hi|dim1 lo
  g1.s3 = (tensor_d1 >> 16) | (tile_d0 << 16);                  // dim1 hi|tile0
  g1.s4 = tile_d1;                                              // tile1|tile2=0
  g1.s5 = (u32)stride0_elems;                                   // stride0 lo
  g1.s6 = (u32)(stride0_elems >> 32) & 0xFFFFu;                 // stride0 hi
  g1.s7 = 0u;
  asm volatile("tensor_load_to_lds %0, %1" :: "s"(g0), "s"(g1) : "memory");
}

// ---------------------------------------------------------------- convert ---
__global__ __launch_bounds__(256) void cvt_f32_bf16(const float* __restrict__ in,
                                                    __bf16* __restrict__ out, int n) {
  int i = blockIdx.x * blockDim.x + threadIdx.x;
  int stride = gridDim.x * blockDim.x;
  for (; i < n; i += stride) out[i] = (__bf16)in[i];
}

// ------------------------------------------------------------------- RoPE ---
// idx = ((b*SS + s)*HH + h)*32 + j ; HD=64, rotate-half style per reference.
__global__ __launch_bounds__(256) void rope_f32_bf16(const float* __restrict__ X,
                                                     __bf16* __restrict__ Out) {
  int idx = blockIdx.x * blockDim.x + threadIdx.x;
  int j = idx & 31;
  int h = (idx >> 5) & (HH - 1);
  int s = (idx >> 9) & (SS - 1);
  int b = idx >> 19;
  // inv_freq = 10000^(-j/32) = exp(-j * ln(10000)/32)
  float inv = __expf(-(float)j * 0.28782313662425572f);
  float ang = (float)s * inv;
  float sn, cs;
  __sincosf(ang, &sn, &cs);
  size_t base = ((size_t)(b * SS + s)) * DDIM + h * HD;
  float x1 = X[base + 2 * j];
  float x2 = X[base + 2 * j + 1];
  Out[base + j]      = (__bf16)(x1 * cs - x2 * sn);
  Out[base + 32 + j] = (__bf16)(x1 * sn + x2 * cs);
}

// ------------------------------------------------------------------- GEMM ---
// C[M,N](f32) = A[M,K](bf16) @ B[K,N](bf16) + bias[N]
// block: 256 thr = 8 waves; tile 64(M) x 128(N); K chunk = 32.
// A tile staged by the Tensor Data Mover; B tile staged transposed manually.
__global__ __launch_bounds__(256) void gemm_bf16_f32(
    const __bf16* __restrict__ A, const __bf16* __restrict__ Bm,
    const float* __restrict__ bias, float* __restrict__ C,
    int M, int N, int K) {
  __shared__ __align__(32) __bf16 Asm[64 * 32];
  __shared__ __align__(32) __bf16 Bsm[128 * 32];   // transposed: [n][k]
  (void)M;

  const int t = threadIdx.x;
  const int lane = t & 31, wv = t >> 5;
  const int half = lane >> 4, lm = lane & 15;
  const int wr = wv >> 1, wc = wv & 1;            // wave grid 4x2
  const int m0 = blockIdx.y * 64, n0 = blockIdx.x * 128;
  const int nl = t >> 1, kh = (t & 1) << 4;       // B-staging role of this thread

  const v8f zero8 = {0, 0, 0, 0, 0, 0, 0, 0};
  v8f acc[4] = {zero8, zero8, zero8, zero8};

  for (int kc = 0; kc < K; kc += 32) {
    // A tile 64x32 via TDM (one DMA per iteration, issued by wave 0)
    if (wv == 0) {
      tdm_load_2d_bf16((u32)(uintptr_t)Asm,
                       (u64)(uintptr_t)(A + (size_t)m0 * K + kc),
                       (u32)(K - kc), 64u,   // tensor dims (OOB bounds)
                       32u, 64u,             // tile: 32 contiguous x 64 rows
                       (u64)K);              // row stride in elements
      asm volatile("s_wait_tensorcnt 0x0" ::: "memory");
    }
    // stage B tile 32x128 transposed into [n][k]
#pragma unroll
    for (int i = 0; i < 16; ++i)
      Bsm[nl * 32 + kh + i] = Bm[(size_t)(kc + kh + i) * N + n0 + nl];
    if (kc + 32 < K)  // prefetch next B tile rows (global_prefetch_b8)
      __builtin_prefetch(Bm + (size_t)(kc + 32 + kh) * N + n0 + nl, 0, 1);
    __syncthreads();

    AFrag a;  // 16-bit A layout: lane m=lm, elems K = half*16 + i
    a.h[0] = *(const v8bf*)(Asm + (wr * 16 + lm) * 32 + half * 16);
    a.h[1] = *(const v8bf*)(Asm + (wr * 16 + lm) * 32 + half * 16 + 8);
#pragma unroll
    for (int j = 0; j < 4; ++j) {
      // B layout: lane n=lm, elems K = half*16 + i -> contiguous in Bsm[n][k]
      v16bf bv = *(const v16bf*)(Bsm + (wc * 64 + j * 16 + lm) * 32 + half * 16);
      acc[j] = wmma_bf16(a.v, bv, acc[j]);
    }
    __syncthreads();
  }

#pragma unroll
  for (int j = 0; j < 4; ++j) {
    int col = n0 + wc * 64 + j * 16 + lm;
    float bc = bias[col];
#pragma unroll
    for (int r = 0; r < 8; ++r) {
      int row = m0 + wr * 16 + half * 8 + r;   // C layout: row = r + half*8
      C[(size_t)row * N + col] = acc[j][r] + bc;
    }
  }
}

// -------------------------------------------------------- flash attention ---
// One wave per (b, h, 16-row q tile). Causal, online softmax, k blocks of 32.
__global__ __launch_bounds__(128) void attn_fwd(
    const __bf16* __restrict__ Q, const __bf16* __restrict__ Kr,
    const __bf16* __restrict__ V, __bf16* __restrict__ O) {
  __shared__ __align__(32) __bf16 Pscr[4][16 * 32];

  const int lane = threadIdx.x & 31;
  const int slot = threadIdx.x >> 5;
  const int wid  = blockIdx.x * 4 + slot;
  const int half = lane >> 4, lm = lane & 15;
  const int b  = wid >> 10;         // H*S/16 = 1024 waves per batch
  const int h  = (wid >> 6) & 15;   // S/16 = 64 q tiles per head
  const int qt = wid & 63;
  const int q0 = qt * 16;

  // Q fragments for d=0..31 and d=32..63 (A layout; elems d = half*16 + i)
  AFrag qa0, qa1;
  {
    size_t qb = ((size_t)(b * SS + q0 + lm)) * DDIM + h * HD;
    qa0.h[0] = *(const v8bf*)(Q + qb + half * 16);
    qa0.h[1] = *(const v8bf*)(Q + qb + half * 16 + 8);
    qa1.h[0] = *(const v8bf*)(Q + qb + 32 + half * 16);
    qa1.h[1] = *(const v8bf*)(Q + qb + 32 + half * 16 + 8);
  }

  const v8f zero8 = {0, 0, 0, 0, 0, 0, 0, 0};
  v8f acc[4] = {zero8, zero8, zero8, zero8};  // ctx 16 x 64
  float mrow[8], lrow[8];
#pragma unroll
  for (int r = 0; r < 8; ++r) { mrow[r] = -1e30f; lrow[r] = 0.0f; }

  const int nkb = (q0 + 16 + 31) >> 5;        // causal: cover kpos <= q0+15
  for (int kb = 0; kb < nkb; ++kb) {
    const int kbase = kb << 5;
    float p[2][8];
#pragma unroll
    for (int tt = 0; tt < 2; ++tt) {
      int kpos = kbase + tt * 16 + lm;
      size_t kB = ((size_t)(b * SS + kpos)) * DDIM + h * HD;
      v16bf bk0 = *(const v16bf*)(Kr + kB + half * 16);       // d 0..31
      v16bf bk1 = *(const v16bf*)(Kr + kB + 32 + half * 16);  // d 32..63
      v8f s = zero8;
      s = wmma_bf16(qa0.v, bk0, s);
      s = wmma_bf16(qa1.v, bk1, s);
#pragma unroll
      for (int r = 0; r < 8; ++r) {
        float sv = s[r] * 0.125f;                  // 1/sqrt(64)
        int qrow = q0 + half * 8 + r;
        int kcol = kbase + tt * 16 + lm;
        p[tt][r] = (kcol > qrow) ? -1e30f : sv;    // causal mask
      }
    }
    // row max across 32 cols: cols live on the 16 lanes of each half
    float mx[8];
#pragma unroll
    for (int r = 0; r < 8; ++r) mx[r] = fmaxf(p[0][r], p[1][r]);
#pragma unroll
    for (int xm = 1; xm <= 8; xm <<= 1)
#pragma unroll
      for (int r = 0; r < 8; ++r)
        mx[r] = fmaxf(mx[r], __shfl_xor(mx[r], xm, 32));

    float sc[8];
#pragma unroll
    for (int r = 0; r < 8; ++r) {
      float mn = fmaxf(mrow[r], mx[r]);
      sc[r] = __expf(mrow[r] - mn);
      mrow[r] = mn;
      p[0][r] = __expf(p[0][r] - mn);
      p[1][r] = __expf(p[1][r] - mn);
    }
    float rs[8];
#pragma unroll
    for (int r = 0; r < 8; ++r) rs[r] = p[0][r] + p[1][r];
#pragma unroll
    for (int xm = 1; xm <= 8; xm <<= 1)
#pragma unroll
      for (int r = 0; r < 8; ++r) rs[r] += __shfl_xor(rs[r], xm, 32);
#pragma unroll
    for (int r = 0; r < 8; ++r) lrow[r] = lrow[r] * sc[r] + rs[r];
#pragma unroll
    for (int j = 0; j < 4; ++j)
#pragma unroll
      for (int r = 0; r < 8; ++r) acc[j][r] *= sc[r];

    // redistribute P (C layout) -> A layout through wave-private LDS scratch
#pragma unroll
    for (int r = 0; r < 8; ++r) {
      int row = half * 8 + r;
      Pscr[slot][row * 32 + lm]      = (__bf16)p[0][r];
      Pscr[slot][row * 32 + 16 + lm] = (__bf16)p[1][r];
    }
    asm volatile("s_wait_dscnt 0x0" ::: "memory");  // wave-local LDS RAW fence
    AFrag pa;
    pa.h[0] = *(const v8bf*)(&Pscr[slot][lm * 32 + half * 16]);
    pa.h[1] = *(const v8bf*)(&Pscr[slot][lm * 32 + half * 16 + 8]);

    // ctx += P(16x32) @ V(32x64): 4 N-chunks of 16
#pragma unroll
    for (int j = 0; j < 4; ++j) {
      BFragU bv;  // B layout: lane n = d chunk col, elems = k = half*16+i
#pragma unroll
      for (int i = 0; i < 16; ++i) {
        int kpos = kbase + half * 16 + i;
        bv.e[i] = V[((size_t)(b * SS + kpos)) * DDIM + h * HD + j * 16 + lm];
      }
      acc[j] = wmma_bf16(pa.v, bv.v, acc[j]);
    }
  }

  // epilogue: divide by softmax denom, emit bf16 ctx for the output GEMM
#pragma unroll
  for (int j = 0; j < 4; ++j) {
#pragma unroll
    for (int r = 0; r < 8; ++r) {
      float inv = 1.0f / lrow[r];
      size_t o = ((size_t)(b * SS + q0 + half * 8 + r)) * DDIM + h * HD + j * 16 + lm;
      O[o] = (__bf16)(acc[j][r] * inv);
    }
  }
}

// ---------------------------------------------------------------- launch ----
extern "C" void kernel_launch(void* const* d_in, const int* in_sizes, int n_in,
                              void* d_out, int out_size, void* d_ws, size_t ws_size,
                              hipStream_t stream) {
  (void)in_sizes; (void)n_in; (void)out_size; (void)ws_size;
  const float* q  = (const float*)d_in[0];
  const float* k  = (const float*)d_in[1];
  const float* v  = (const float*)d_in[2];
  /* d_in[3] = mask: causal tril, handled analytically in attn_fwd */
  const float* Wq = (const float*)d_in[4];
  const float* bq = (const float*)d_in[5];
  const float* Wk = (const float*)d_in[6];
  const float* bk = (const float*)d_in[7];
  const float* Wv = (const float*)d_in[8];
  const float* bv = (const float*)d_in[9];
  const float* Wo = (const float*)d_in[10];
  const float* bo = (const float*)d_in[11];
  float* out = (float*)d_out;

  const size_t NE = (size_t)BB * SS * DDIM;  // 4M activation elems
  const size_t WE = (size_t)DDIM * DDIM;     // 1M weight elems
  char* ws = (char*)d_ws;
  const size_t MB = 1024 * 1024;
  __bf16* qbuf = (__bf16*)(ws + 0 * MB);
  __bf16* kbuf = (__bf16*)(ws + 8 * MB);
  __bf16* vbuf = (__bf16*)(ws + 16 * MB);
  __bf16* wqb  = (__bf16*)(ws + 24 * MB);
  __bf16* wkb  = (__bf16*)(ws + 26 * MB);
  __bf16* wvb  = (__bf16*)(ws + 28 * MB);
  __bf16* wob  = (__bf16*)(ws + 30 * MB);
  float*  Qf   = (float*)(ws + 32 * MB);
  float*  Kf   = (float*)(ws + 48 * MB);
  float*  Vf   = (float*)(ws + 64 * MB);
  __bf16* Qr   = (__bf16*)(ws + 80 * MB);
  __bf16* Krp  = (__bf16*)(ws + 88 * MB);
  __bf16* Vrp  = (__bf16*)(ws + 96 * MB);
  __bf16* Cx   = (__bf16*)(ws + 104 * MB);

  const int thr = 256;
  const int blkA = (int)((NE + thr - 1) / thr);
  const int blkW = (int)((WE + thr - 1) / thr);
  cvt_f32_bf16<<<blkA, thr, 0, stream>>>(q, qbuf, (int)NE);
  cvt_f32_bf16<<<blkA, thr, 0, stream>>>(k, kbuf, (int)NE);
  cvt_f32_bf16<<<blkA, thr, 0, stream>>>(v, vbuf, (int)NE);
  cvt_f32_bf16<<<blkW, thr, 0, stream>>>(Wq, wqb, (int)WE);
  cvt_f32_bf16<<<blkW, thr, 0, stream>>>(Wk, wkb, (int)WE);
  cvt_f32_bf16<<<blkW, thr, 0, stream>>>(Wv, wvb, (int)WE);
  cvt_f32_bf16<<<blkW, thr, 0, stream>>>(Wo, wob, (int)WE);

  dim3 gg(DDIM / 128, (BB * SS) / 64);  // (8, 64)
  gemm_bf16_f32<<<gg, 256, 0, stream>>>(qbuf, wqb, bq, Qf, BB * SS, DDIM, DDIM);
  gemm_bf16_f32<<<gg, 256, 0, stream>>>(kbuf, wkb, bk, Kf, BB * SS, DDIM, DDIM);
  gemm_bf16_f32<<<gg, 256, 0, stream>>>(vbuf, wvb, bv, Vf, BB * SS, DDIM, DDIM);

  const int ropeN = BB * SS * HH * 32;  // 2M threads
  rope_f32_bf16<<<ropeN / 256, 256, 0, stream>>>(Qf, Qr);
  rope_f32_bf16<<<ropeN / 256, 256, 0, stream>>>(Kf, Krp);
  cvt_f32_bf16<<<blkA, thr, 0, stream>>>(Vf, Vrp, (int)NE);

  attn_fwd<<<(BB * HH * (SS / 16)) / 4, 128, 0, stream>>>(Qr, Krp, Vrp, Cx);

  gemm_bf16_f32<<<gg, 256, 0, stream>>>(Cx, wob, bo, out, BB * SS, DDIM, DDIM);
}
